// DeformableConv_4896262718062
// MI455X (gfx1250) — compile-verified, matrix operand order checked by reading, not run on hardware
//
#include <hip/hip_runtime.h>

// ---------------------------------------------------------------------------
// Deformable conv on MI455X (gfx1250, wave32):
//   out[b,oc,y,x] = sum_{c,j,i} bilinear(feature, offset-grid) * w[oc,c,j,i]
// == GEMM  C[M=256, N=B*H*W=100352] = A[M,K=1152] * S[K,N]
//
// v2: each block now computes the FULL M=256 for its 128-pixel tile, so the
// expensive part (scattered 4B bilinear gathers, ~L2-resident) is done once
// instead of once per 128-row M-tile. Per wave: 8(M) x 2(N) WMMA tiles,
// 16 v_wmma_f32_16x16x32_f16 per K-step. S is never materialized in HBM.
// ---------------------------------------------------------------------------

typedef _Float16 f16;
typedef f16  v16h __attribute__((ext_vector_type(16)));
typedef float v8f __attribute__((ext_vector_type(8)));

#define KSZ   3
#define CIN   128
#define HH    56
#define WW    56
#define HW    (HH * WW)          // 3136
#define BB    32
#define OUTC  256
#define KTOT  (CIN * 9)          // 1152
#define NTOT  (BB * HW)          // 100352
#define LDSROW 40                // 32 f16 data + 8 f16 pad = 80B row stride

// --- prep: conv_weight f32 [OUTC,CIN,3,3] -> f16 A-matrix [M=256][K=1152] ---
// k = (j*3+i)*128 + c, row-major in k so GEMM A-tile loads are contiguous.
__global__ __launch_bounds__(256)
void dcn_wconv_kernel(const float* __restrict__ w, f16* __restrict__ wh) {
    int t = blockIdx.x * blockDim.x + threadIdx.x;
    if (t >= OUTC * KTOT) return;
    int m   = t / KTOT;
    int k   = t - m * KTOT;
    int tap = k >> 7;            // k / 128
    int c   = k & 127;
    int j   = tap / 3, i = tap - 3 * j;
    wh[t] = (f16)w[((m * CIN + c) * KSZ + j) * KSZ + i];
}

// --- fused sample + GEMM ----------------------------------------------------
__global__ __launch_bounds__(256)
void dcn_gemm_kernel(const float* __restrict__ feat,
                     const float* __restrict__ off,
                     const f16*  __restrict__ wh,
                     float*      __restrict__ out) {
    __shared__ __align__(16) f16 sA[256 * LDSROW];   // 20.5 KB weight tile
    __shared__ __align__(16) f16 sB[128 * LDSROW];   // 10.2 KB sample tile
    __shared__ float4 sW[128 * 9];                   // bilinear weights
    __shared__ int2   sXY[128 * 9];                  // bilinear int corners

    const int tid = threadIdx.x;
    const int n0  = blockIdx.x * 128;                // pixel-tile base

    // Phase 0: bilinear coefficients for this block's 128 pixels x 9 taps.
    // Reference quirk: offset indexed [b, tap*2+comp, x, y] (x,y swapped).
    for (int t = tid; t < 128 * 9; t += 256) {
        int nl  = t / 9, tap = t - 9 * nl;
        int n   = n0 + nl;
        int b   = n / HW;
        int pix = n - b * HW;
        int y   = pix / WW, x = pix - y * WW;
        const float halfw = (WW - 1) * 0.5f;
        const float halfh = (HH - 1) * 0.5f;
        float ax = ((float)x - halfw) / halfw;
        float ay = ((float)y - halfh) / halfh;
        int   ob = ((b * 18 + tap * 2) * HH + x) * WW + y;
        float gx = off[ob]      + ax;
        float gy = off[ob + HW] + ay;                 // next channel = +H*W
        float ix = ((gx + 1.f) * WW - 1.f) * 0.5f;    // align_corners=False
        float iy = ((gy + 1.f) * HH - 1.f) * 0.5f;
        float ix0 = floorf(ix), iy0 = floorf(iy);
        float fx = ix - ix0,  fy = iy - iy0;
        sW[t]  = make_float4((1.f - fx) * (1.f - fy), fx * (1.f - fy),
                             (1.f - fx) * fy,         fx * fy);
        sXY[t] = make_int2((int)ix0, (int)iy0);
    }

    const int wave  = tid >> 5;
    const int lane  = tid & 31;
    const int l16   = lane & 15;
    const int half  = lane >> 4;
    const int waveM = wave & 1;      // which 128-row half of M=256
    const int waveN = wave >> 1;     // which 32-col quarter of the 128 N-tile

    v8f acc[8][2] = {};

    const int rl = tid >> 1;         // 0..127: staged pixel row for B
    const int kh = (tid & 1) * 16;   // low/high 16 of the 32-wide K slice

    for (int kt = 0; kt < KTOT / 32; ++kt) {
        __syncthreads();             // previous iteration's LDS reads done

        // ---- stage A tile: 256 rows x 32 k of f16 weights (coalesced) ----
        // one full 64B row per thread
        {
            const uint4* g = (const uint4*)(wh + (size_t)tid * KTOT + kt * 32);
            uint4* s = (uint4*)(sA + tid * LDSROW);
            s[0] = g[0];
            s[1] = g[1];
            s[2] = g[2];
            s[3] = g[3];
        }

        // ---- stage B tile: bilinear-sample 16 channels for pixel rl ----
        {
            int tap   = kt >> 2;                       // 128 channels per tap
            int cbase = ((kt & 3) << 5) + kh;          // (kt%4)*32 + kh
            int cidx  = rl * 9 + tap;
            float4 wg = sW[cidx];
            int2   xy = sXY[cidx];
            int n = n0 + rl;
            int b = n / HW;
            const float* fb = feat + (size_t)b * CIN * HW;
            int x0 = xy.x, y0 = xy.y, x1 = x0 + 1, y1 = y0 + 1;
            bool x0k = (x0 >= 0) & (x0 < WW), x1k = (x1 >= 0) & (x1 < WW);
            bool y0k = (y0 >= 0) & (y0 < HH), y1k = (y1 >= 0) & (y1 < HH);
            int x0c = min(max(x0, 0), WW - 1), x1c = min(max(x1, 0), WW - 1);
            int y0c = min(max(y0, 0), HH - 1), y1c = min(max(y1, 0), HH - 1);
            int o00 = y0c * WW + x0c, o01 = y0c * WW + x1c;
            int o10 = y1c * WW + x0c, o11 = y1c * WW + x1c;
            float m00 = (x0k & y0k) ? wg.x : 0.f;
            float m01 = (x1k & y0k) ? wg.y : 0.f;
            float m10 = (x0k & y1k) ? wg.z : 0.f;
            float m11 = (x1k & y1k) ? wg.w : 0.f;
            // prefetch next channel group (L2 temporal) -> global_prefetch_b8
            if (cbase + 16 < CIN)
                __builtin_prefetch(fb + (size_t)(cbase + 16) * HW + o00, 0, 3);
            __align__(16) f16 vals[16];
            #pragma unroll
            for (int cc = 0; cc < 16; ++cc) {
                const float* fc = fb + (size_t)(cbase + cc) * HW;
                float v = m00 * fc[o00] + m01 * fc[o01]
                        + m10 * fc[o10] + m11 * fc[o11];
                vals[cc] = (f16)v;
            }
            uint4* s = (uint4*)(sB + rl * LDSROW + kh);
            s[0] = *(const uint4*)&vals[0];
            s[1] = *(const uint4*)&vals[8];
        }
        __syncthreads();

        // ---- fragment loads (two ds_load_b128 each) + 16 WMMAs ----
        // 16-bit A layout: lane=(l16,half): elems 0..7 -> k=half*8+e,
        //                  elems 8..15 -> k=16+half*8+(e-8)  (ISA 7.12.2)
        v16h bfrag[2];
        #pragma unroll
        for (int ni = 0; ni < 2; ++ni) {
            const f16* p = sB + (waveN * 32 + ni * 16 + l16) * LDSROW + half * 8;
            uint4* q = (uint4*)&bfrag[ni];
            q[0] = *(const uint4*)p;
            q[1] = *(const uint4*)(p + 16);
        }
        #pragma unroll
        for (int mi = 0; mi < 8; ++mi) {
            v16h afrag;
            const f16* p = sA + (waveM * 128 + mi * 16 + l16) * LDSROW + half * 8;
            uint4* q = (uint4*)&afrag;
            q[0] = *(const uint4*)p;
            q[1] = *(const uint4*)(p + 16);
            #pragma unroll
            for (int ni = 0; ni < 2; ++ni)
                acc[mi][ni] = __builtin_amdgcn_wmma_f32_16x16x32_f16(
                    false, afrag, false, bfrag[ni],
                    (short)0, acc[mi][ni], false, false);
        }
    }

    // ---- store: C/D layout VGPR r -> m=(half*8+r), n=l16 (ISA 7.12.2) ----
    #pragma unroll
    for (int ni = 0; ni < 2; ++ni) {
        int n   = n0 + waveN * 32 + ni * 16 + l16;
        int b   = n / HW;
        int pix = n - b * HW;
        #pragma unroll
        for (int mi = 0; mi < 8; ++mi) {
            #pragma unroll
            for (int r = 0; r < 8; ++r) {
                int m = waveM * 128 + mi * 16 + half * 8 + r;
                out[(size_t)(b * OUTC + m) * HW + pix] = acc[mi][ni][r];
            }
        }
    }
}

extern "C" void kernel_launch(void* const* d_in, const int* in_sizes, int n_in,
                              void* d_out, int out_size, void* d_ws, size_t ws_size,
                              hipStream_t stream) {
    const float* feature     = (const float*)d_in[0];   // [32,128,56,56] f32
    const float* offset      = (const float*)d_in[1];   // [32,18,56,56]  f32
    const float* conv_weight = (const float*)d_in[2];   // [256,128,3,3]  f32
    float*       out         = (float*)d_out;           // [32,256,56,56] f32
    f16*         wh          = (f16*)d_ws;              // 256*1152 f16 = 576KB

    int wtot = OUTC * KTOT;
    dcn_wconv_kernel<<<(wtot + 255) / 256, 256, 0, stream>>>(conv_weight, wh);

    dim3 grid(NTOT / 128);                              // 784 blocks, full M
    dcn_gemm_kernel<<<grid, 256, 0, stream>>>(feature, offset, wh, out);
}
